// SparseAutoencoder_4518305596079
// MI455X (gfx1250) — matrix operand order repaired and checked by reading, not compiled, not run on hardware
//
#include <hip/hip_runtime.h>
#include <hip/hip_bf16.h>

#define D_MODEL   4096
#define D_HIDDEN  16384
#define N_TOKENS  4096
#define TOPK      128
#define LN_EPS    1e-5f

typedef __attribute__((ext_vector_type(16))) __bf16 v16bf;
typedef __attribute__((ext_vector_type(8)))  __bf16 v8bf;
typedef __attribute__((ext_vector_type(8)))  float  v8f;

union V16 { v16bf v; v8bf h[2]; };

// CDNA5 async global->LDS path (ASYNCcnt), if the toolchain exposes it.
#if defined(__HIP_DEVICE_COMPILE__) && defined(__has_builtin)
#if __has_builtin(__builtin_amdgcn_global_load_async_to_lds_b128) && \
    __has_builtin(__builtin_amdgcn_s_wait_asynccnt)
#define USE_ASYNC_LDS 1
#endif
#endif

// exact param types per hipcc diagnostic: (AS1 int4*, AS3 int4*, imm, imm)
typedef int v4i __attribute__((vector_size(16)));
typedef __attribute__((address_space(1))) v4i* g_v4i_ptr;
typedef __attribute__((address_space(3))) v4i* l_v4i_ptr;

__device__ __forceinline__ __bf16 f2bf(float f) {
  unsigned u = __builtin_bit_cast(unsigned, f);
  unsigned short s = (unsigned short)((u + 0x7FFFu + ((u >> 16) & 1u)) >> 16);
  return __builtin_bit_cast(__bf16, s);
}
__device__ __forceinline__ float bf2f(__bf16 h) {
  unsigned short s = __builtin_bit_cast(unsigned short, h);
  unsigned u = ((unsigned)s) << 16;
  return __builtin_bit_cast(float, u);
}
// monotonic float->uint key: larger float => larger key
__device__ __forceinline__ unsigned fkey(float f) {
  unsigned u = __builtin_bit_cast(unsigned, f);
  return u ^ ((unsigned)((int)u >> 31) | 0x80000000u);
}

__device__ __forceinline__ void cp16_g2l(void* lds_dst, const void* gsrc) {
#ifdef USE_ASYNC_LDS
  __builtin_amdgcn_global_load_async_to_lds_b128(
      (g_v4i_ptr)(void*)gsrc, (l_v4i_ptr)lds_dst, 0, 0);
#else
  *(float4*)lds_dst = *(const float4*)gsrc;
#endif
}

// ---------------------------------------------------------------------------
// 1) LayerNorm (unbiased std) + subtract b_pre, split result to bf16 hi/lo
// ---------------------------------------------------------------------------
__global__ void __launch_bounds__(256) sae_ln_split(
    const float* __restrict__ x, const float* __restrict__ b_pre,
    __bf16* __restrict__ xh, __bf16* __restrict__ xl,
    float* __restrict__ mus, float* __restrict__ stds)
{
  __shared__ float rs[256], rq[256];
  const int t = threadIdx.x;
  const int row = blockIdx.x;
  const float* xp = x + (size_t)row * D_MODEL;

  float s = 0.f, q = 0.f;
  for (int i = t; i < D_MODEL; i += 256) { float f = xp[i]; s += f; q += f * f; }
  rs[t] = s; rq[t] = q;
  __syncthreads();
  for (int off = 128; off > 0; off >>= 1) {
    if (t < off) { rs[t] += rs[t + off]; rq[t] += rq[t + off]; }
    __syncthreads();
  }
  const float mu = rs[0] * (1.f / (float)D_MODEL);
  float var = (rq[0] - (float)D_MODEL * mu * mu) * (1.f / (float)(D_MODEL - 1));
  var = fmaxf(var, 0.f);
  const float sd = sqrtf(var);
  const float inv = 1.f / (sd + LN_EPS);

  __bf16* xhp = xh + (size_t)row * D_MODEL;
  __bf16* xlp = xl + (size_t)row * D_MODEL;
  for (int i = t; i < D_MODEL; i += 256) {
    float xn = (xp[i] - mu) * inv - b_pre[i];
    __bf16 hi = f2bf(xn);
    __bf16 lo = f2bf(xn - bf2f(hi));
    xhp[i] = hi; xlp[i] = lo;
  }
  if (t == 0) { mus[row] = mu; stds[row] = sd; }
}

// ---------------------------------------------------------------------------
// 2) Encoder GEMM: pre = xn @ w_enc + b_enc   via bf16 WMMA with hi/lo split
//    Block tile: 128(M) x 64(N), K-step 32.  8 waves -> each wave 32x32 out.
//    1D grid with L2-aware swizzle: M-tile fast inside 32-panel chunks, so
//    concurrently-resident blocks share both the A slice and the w_enc panel.
// ---------------------------------------------------------------------------
__global__ void __launch_bounds__(256) sae_enc_gemm(
    const __bf16* __restrict__ xh, const __bf16* __restrict__ xl,
    const float* __restrict__ w_enc, const float* __restrict__ b_enc,
    float* __restrict__ pre)
{
  __shared__ __bf16 sAh[128 * 40];   // [row][k], pad stride 40 keeps 16B align
  __shared__ __bf16 sAl[128 * 40];
  __shared__ __bf16 sBh[64 * 40];    // transposed: [n][k]
  __shared__ __bf16 sBl[64 * 40];

  const int t    = threadIdx.x;
  const int lane = t & 31;
  const int wave = t >> 5;
  const int wm   = wave >> 1;     // 0..3 : M sub-tile
  const int wn   = wave & 1;      // 0..1 : N sub-tile
  const int half = lane >> 4;     // 0/1 lane half
  const int lr   = lane & 15;

  // grid swizzle: 8192 blocks = 8 chunks x (32 panels x 32 M-tiles), M fast
  const int bx    = blockIdx.x;
  const int chunk = bx >> 10;            // 0..7
  const int r     = bx & 1023;
  const int m0    = (r & 31) * 128;      // M-tile (fast-varying)
  const int n0    = ((chunk << 5) + (r >> 5)) * 64;  // w_enc panel

  // A staging: 512 16B chunks per array / 256 threads = 2 each
  const int ca0 = t, ca1 = t + 256;
  // B staging: 8 consecutive floats per thread
  const int kr = t >> 3;          // 0..31
  const int nb = (t & 7) * 8;     // 0..56

  v8f acc00 = {0.f,0.f,0.f,0.f,0.f,0.f,0.f,0.f};
  v8f acc01 = acc00, acc10 = acc00, acc11 = acc00;

  for (int k0 = 0; k0 < D_MODEL; k0 += 32) {
    // ---- stage A tiles (bf16 hi/lo in ws): async global->LDS if available
    {
      int r0 = ca0 >> 2, kc0 = (ca0 & 3) * 8;
      int r1 = ca1 >> 2, kc1 = (ca1 & 3) * 8;
      size_t g0 = (size_t)(m0 + r0) * D_MODEL + k0 + kc0;
      size_t g1 = (size_t)(m0 + r1) * D_MODEL + k0 + kc1;
      cp16_g2l(sAh + r0 * 40 + kc0, xh + g0);
      cp16_g2l(sAh + r1 * 40 + kc1, xh + g1);
      cp16_g2l(sAl + r0 * 40 + kc0, xl + g0);
      cp16_g2l(sAl + r1 * 40 + kc1, xl + g1);
    }
    // ---- stage B tile: fp32 -> bf16 hi/lo, transposed to [n][k] ----
    {
      const float* src = w_enc + (size_t)(k0 + kr) * D_HIDDEN + n0 + nb;
      float4 f0 = *(const float4*)src;
      float4 f1 = *(const float4*)(src + 4);
      float fv[8] = {f0.x, f0.y, f0.z, f0.w, f1.x, f1.y, f1.z, f1.w};
#pragma unroll
      for (int j = 0; j < 8; ++j) {
        __bf16 hi = f2bf(fv[j]);
        __bf16 lo = f2bf(fv[j] - bf2f(hi));
        sBh[(nb + j) * 40 + kr] = hi;
        sBl[(nb + j) * 40 + kr] = lo;
      }
      if (k0 + 32 < D_MODEL)
        __builtin_prefetch(src + 32 * D_HIDDEN, 0, 1);  // next K tile of w_enc
    }
#ifdef USE_ASYNC_LDS
    __builtin_amdgcn_s_wait_asynccnt(0);   // our async LDS writes landed
#endif
    __syncthreads();

    // ---- fragments per ISA 16-bit A(16x32)/B(32x16) wave32 layouts ----
    V16 ah0, ah1, al0, al1, bh0, bh1, bl0, bl1;
    {
      const int ka  = half * 8;
      const int rA0 = wm * 32 + lr;
      const int rA1 = wm * 32 + 16 + lr;
      ah0.h[0] = *(const v8bf*)(sAh + rA0 * 40 + ka);
      ah0.h[1] = *(const v8bf*)(sAh + rA0 * 40 + ka + 16);
      al0.h[0] = *(const v8bf*)(sAl + rA0 * 40 + ka);
      al0.h[1] = *(const v8bf*)(sAl + rA0 * 40 + ka + 16);
      ah1.h[0] = *(const v8bf*)(sAh + rA1 * 40 + ka);
      ah1.h[1] = *(const v8bf*)(sAh + rA1 * 40 + ka + 16);
      al1.h[0] = *(const v8bf*)(sAl + rA1 * 40 + ka);
      al1.h[1] = *(const v8bf*)(sAl + rA1 * 40 + ka + 16);
      const int kb  = half * 8;
      const int cB0 = wn * 32 + lr;
      const int cB1 = wn * 32 + 16 + lr;
      bh0.h[0] = *(const v8bf*)(sBh + cB0 * 40 + kb);
      bh0.h[1] = *(const v8bf*)(sBh + cB0 * 40 + kb + 16);
      bl0.h[0] = *(const v8bf*)(sBl + cB0 * 40 + kb);
      bl0.h[1] = *(const v8bf*)(sBl + cB0 * 40 + kb + 16);
      bh1.h[0] = *(const v8bf*)(sBh + cB1 * 40 + kb);
      bh1.h[1] = *(const v8bf*)(sBh + cB1 * 40 + kb + 16);
      bl1.h[0] = *(const v8bf*)(sBl + cB1 * 40 + kb);
      bl1.h[1] = *(const v8bf*)(sBl + cB1 * 40 + kb + 16);
    }
    // split-compensated product: Ah*Bh + Ah*Bl + Al*Bh
    acc00 = __builtin_amdgcn_wmma_f32_16x16x32_bf16(false, ah0.v, false, bh0.v, (short)0, acc00, false, false);
    acc00 = __builtin_amdgcn_wmma_f32_16x16x32_bf16(false, ah0.v, false, bl0.v, (short)0, acc00, false, false);
    acc00 = __builtin_amdgcn_wmma_f32_16x16x32_bf16(false, al0.v, false, bh0.v, (short)0, acc00, false, false);

    acc01 = __builtin_amdgcn_wmma_f32_16x16x32_bf16(false, ah0.v, false, bh1.v, (short)0, acc01, false, false);
    acc01 = __builtin_amdgcn_wmma_f32_16x16x32_bf16(false, ah0.v, false, bl1.v, (short)0, acc01, false, false);
    acc01 = __builtin_amdgcn_wmma_f32_16x16x32_bf16(false, al0.v, false, bh1.v, (short)0, acc01, false, false);

    acc10 = __builtin_amdgcn_wmma_f32_16x16x32_bf16(false, ah1.v, false, bh0.v, (short)0, acc10, false, false);
    acc10 = __builtin_amdgcn_wmma_f32_16x16x32_bf16(false, ah1.v, false, bl0.v, (short)0, acc10, false, false);
    acc10 = __builtin_amdgcn_wmma_f32_16x16x32_bf16(false, al1.v, false, bh0.v, (short)0, acc10, false, false);

    acc11 = __builtin_amdgcn_wmma_f32_16x16x32_bf16(false, ah1.v, false, bh1.v, (short)0, acc11, false, false);
    acc11 = __builtin_amdgcn_wmma_f32_16x16x32_bf16(false, ah1.v, false, bl1.v, (short)0, acc11, false, false);
    acc11 = __builtin_amdgcn_wmma_f32_16x16x32_bf16(false, al1.v, false, bh1.v, (short)0, acc11, false, false);
    __syncthreads();
  }

  // ---- epilogue: C layout -> VGPR r holds (M = base + r + 8*half, N = base + lr)
  {
    const int nc0 = n0 + wn * 32 + lr;
    const float be0 = b_enc[nc0];
    const float be1 = b_enc[nc0 + 16];
    const int mr0 = m0 + wm * 32 + half * 8;
#pragma unroll
    for (int r8 = 0; r8 < 8; ++r8) {
      pre[(size_t)(mr0 + r8) * D_HIDDEN + nc0]           = acc00[r8] + be0;
      pre[(size_t)(mr0 + r8) * D_HIDDEN + nc0 + 16]      = acc01[r8] + be1;
      pre[(size_t)(mr0 + 16 + r8) * D_HIDDEN + nc0]      = acc10[r8] + be0;
      pre[(size_t)(mr0 + 16 + r8) * D_HIDDEN + nc0 + 16] = acc11[r8] + be1;
    }
  }
}

// ---------------------------------------------------------------------------
// 3) Exact per-row top-k (radix select, index-ordered ties), in place; also
//    emits compact (idx, relu(val)) lists for the sparse decoder.
// ---------------------------------------------------------------------------
__global__ void __launch_bounds__(256) sae_topk(
    float* __restrict__ lat, int* __restrict__ nzi, float* __restrict__ nzv)
{
  __shared__ unsigned hist[256];
  __shared__ unsigned sEq[256];
  __shared__ unsigned sSl[256];
  __shared__ unsigned sSel, sWant;

  const int t = threadIdx.x;
  const int row = blockIdx.x;
  float* rowp = lat + (size_t)row * D_HIDDEN;

  unsigned prefix = 0, prefmask = 0, want = TOPK;

  for (int shift = 24; shift >= 0; shift -= 8) {
    hist[t] = 0;
    __syncthreads();
    for (int i = t; i < D_HIDDEN; i += 256) {
      unsigned u = fkey(rowp[i]);
      if ((u & prefmask) == prefix)
        atomicAdd(&hist[(u >> shift) & 255u], 1u);
    }
    __syncthreads();
    if (t == 0) {
      unsigned cum = 0, sel = 0, rem = want;
      for (int b = 255; b >= 0; --b) {
        unsigned c = hist[b];
        if (cum + c >= want) { sel = (unsigned)b; rem = want - cum; break; }
        cum += c;
      }
      sSel = sel; sWant = rem;
    }
    __syncthreads();
    prefix   |= sSel << shift;
    want      = sWant;
    prefmask |= (0xFFu << shift);
    __syncthreads();
  }

  const unsigned T = prefix;                 // key of the k-th largest element
  const int base = t * (D_HIDDEN / 256);     // contiguous chunk of 64
  unsigned eqc = 0, gtc = 0;
  for (int i = 0; i < D_HIDDEN / 256; ++i) {
    unsigned u = fkey(rowp[base + i]);
    if (u == T) ++eqc; else if (u > T) ++gtc;
  }
  sEq[t] = eqc;
  __syncthreads();
  if (t == 0) { unsigned run = 0; for (int i = 0; i < 256; ++i) { unsigned c = sEq[i]; sEq[i] = run; run += c; } }
  __syncthreads();
  const unsigned eqBase = sEq[t];
  const unsigned eqSel  = (eqBase >= want) ? 0u : ((want - eqBase) < eqc ? (want - eqBase) : eqc);
  sSl[t] = gtc + eqSel;
  __syncthreads();
  if (t == 0) { unsigned run = 0; for (int i = 0; i < 256; ++i) { unsigned c = sSl[i]; sSl[i] = run; run += c; } }
  __syncthreads();

  unsigned slot  = sSl[t];
  unsigned eqIdx = eqBase;
  for (int i = 0; i < D_HIDDEN / 256; ++i) {
    float f = rowp[base + i];
    unsigned u = fkey(f);
    bool sel;
    if (u == T) { sel = (eqIdx < want); ++eqIdx; }
    else        { sel = (u > T); }
    float ov = sel ? fmaxf(f, 0.f) : 0.f;
    rowp[base + i] = ov;
    if (sel) {
      nzi[(size_t)row * TOPK + slot] = base + i;
      nzv[(size_t)row * TOPK + slot] = ov;
      ++slot;
    }
  }
}

// ---------------------------------------------------------------------------
// 4) Sparse decode: recons = (sum_j v_j * w_dec[idx_j,:] + b_pre) * std + mu
// ---------------------------------------------------------------------------
__global__ void __launch_bounds__(256) sae_decode(
    const int* __restrict__ nzi, const float* __restrict__ nzv,
    const float* __restrict__ w_dec, const float* __restrict__ b_pre,
    const float* __restrict__ mus, const float* __restrict__ stds,
    float* __restrict__ recons)
{
  __shared__ int   si[TOPK];
  __shared__ float sv[TOPK];
  const int t = threadIdx.x;
  const int row = blockIdx.x;
  if (t < TOPK) {
    si[t] = nzi[(size_t)row * TOPK + t];
    sv[t] = nzv[(size_t)row * TOPK + t];
  }
  __syncthreads();

  float4 a0 = {0,0,0,0}, a1 = {0,0,0,0}, a2 = {0,0,0,0}, a3 = {0,0,0,0};
  for (int j = 0; j < TOPK; ++j) {
    const float4* wr = (const float4*)(w_dec + (size_t)si[j] * D_MODEL);
    const float v = sv[j];
    float4 w0 = wr[t], w1 = wr[t + 256], w2 = wr[t + 512], w3 = wr[t + 768];
    a0.x += v * w0.x; a0.y += v * w0.y; a0.z += v * w0.z; a0.w += v * w0.w;
    a1.x += v * w1.x; a1.y += v * w1.y; a1.z += v * w1.z; a1.w += v * w1.w;
    a2.x += v * w2.x; a2.y += v * w2.y; a2.z += v * w2.z; a2.w += v * w2.w;
    a3.x += v * w3.x; a3.y += v * w3.y; a3.z += v * w3.z; a3.w += v * w3.w;
  }
  const float s = stds[row], m = mus[row];
  const float4* bp = (const float4*)b_pre;
  float4* out = (float4*)(recons + (size_t)row * D_MODEL);
  float4 b0 = bp[t], b1 = bp[t + 256], b2 = bp[t + 512], b3 = bp[t + 768];
  float4 r0, r1, r2, r3;
  r0.x = (a0.x + b0.x) * s + m; r0.y = (a0.y + b0.y) * s + m;
  r0.z = (a0.z + b0.z) * s + m; r0.w = (a0.w + b0.w) * s + m;
  r1.x = (a1.x + b1.x) * s + m; r1.y = (a1.y + b1.y) * s + m;
  r1.z = (a1.z + b1.z) * s + m; r1.w = (a1.w + b1.w) * s + m;
  r2.x = (a2.x + b2.x) * s + m; r2.y = (a2.y + b2.y) * s + m;
  r2.z = (a2.z + b2.z) * s + m; r2.w = (a2.w + b2.w) * s + m;
  r3.x = (a3.x + b3.x) * s + m; r3.y = (a3.y + b3.y) * s + m;
  r3.z = (a3.z + b3.z) * s + m; r3.w = (a3.w + b3.w) * s + m;
  out[t] = r0; out[t + 256] = r1; out[t + 512] = r2; out[t + 768] = r3;
}

// ---------------------------------------------------------------------------
extern "C" void kernel_launch(void* const* d_in, const int* in_sizes, int n_in,
                              void* d_out, int out_size, void* d_ws, size_t ws_size,
                              hipStream_t stream) {
  (void)in_sizes; (void)n_in; (void)out_size; (void)ws_size;

  const float* x     = (const float*)d_in[0];
  const float* w_enc = (const float*)d_in[1];
  const float* w_dec = (const float*)d_in[2];
  const float* b_enc = (const float*)d_in[3];
  const float* b_pre = (const float*)d_in[4];
  // d_in[5] is k (always 128 for this module)

  float* recons = (float*)d_out;
  float* lat    = (float*)d_out + (size_t)N_TOKENS * D_MODEL;   // pre_acts -> latents in place

  char* w = (char*)d_ws;
  __bf16* xh  = (__bf16*)w;  w += (size_t)N_TOKENS * D_MODEL * 2;
  __bf16* xl  = (__bf16*)w;  w += (size_t)N_TOKENS * D_MODEL * 2;
  float* mus  = (float*)w;   w += (size_t)N_TOKENS * 4;
  float* stds = (float*)w;   w += (size_t)N_TOKENS * 4;
  int*   nzi  = (int*)w;     w += (size_t)N_TOKENS * TOPK * 4;
  float* nzv  = (float*)w;   w += (size_t)N_TOKENS * TOPK * 4;

  sae_ln_split<<<N_TOKENS, 256, 0, stream>>>(x, b_pre, xh, xl, mus, stds);
  sae_enc_gemm<<<(D_HIDDEN / 64) * (N_TOKENS / 128), 256, 0, stream>>>(xh, xl, w_enc, b_enc, lat);
  sae_topk<<<N_TOKENS, 256, 0, stream>>>(lat, nzi, nzv);
  sae_decode<<<N_TOKENS, 256, 0, stream>>>(nzi, nzv, w_dec, b_pre, mus, stds, recons);
}
